// DeformableAttention_83743272337538
// MI455X (gfx1250) — matrix operand order repaired and checked by reading, not compiled, not run on hardware
//
#include <hip/hip_runtime.h>
#include <hip/hip_bf16.h>
#include <math.h>

// ---------------------------------------------------------------------------
// Problem constants (from the reference)
// ---------------------------------------------------------------------------
static constexpr int BQ = 2;
static constexpr int LQ = 8192;
static constexpr int EMB = 256;     // N_HEADS * D_HEAD; also K of every GEMM
static constexpr int NH  = 8;
static constexpr int NP  = 8;
static constexpr int DH  = 32;
static constexpr int M   = BQ * LQ; // 16384 rows of activations

// ---------------------------------------------------------------------------
// Types for WMMA fragments (CDNA5 wave32, V_WMMA_F32_16X16X32_BF16)
// ---------------------------------------------------------------------------
typedef __attribute__((ext_vector_type(16))) __bf16       v16bf;
typedef __attribute__((ext_vector_type(8)))  float        v8f;
typedef __attribute__((ext_vector_type(4)))  unsigned int u32x4;

union ABfrag {
    v16bf v;
    u32x4 u[2];
};

// ---------------------------------------------------------------------------
// fp32 -> bf16 elementwise convert
// ---------------------------------------------------------------------------
__global__ void cvt_f32_to_bf16(const float* __restrict__ in,
                                __bf16* __restrict__ out, int n) {
    int i = blockIdx.x * blockDim.x + threadIdx.x;
    if (i < n) out[i] = (__bf16)in[i];
}

// ---------------------------------------------------------------------------
// Weight prep: W [K x N] fp32 row-major -> WT [N x K] bf16 row-major
// (so the WMMA B operand loads contiguous K per lane)
// ---------------------------------------------------------------------------
__global__ void transpose_cvt(const float* __restrict__ W,
                              __bf16* __restrict__ WT, int K, int N) {
    int i = blockIdx.x * blockDim.x + threadIdx.x;
    if (i < K * N) {
        int k = i / N;
        int n = i - k * N;
        WT[n * K + k] = (__bf16)W[k * N + n];
    }
}

// ---------------------------------------------------------------------------
// Fold the two value projections into one effective weight/bias:
//   Weff = Wv_out @ Wv_in            (stored transposed, bf16, [N x K])
//   beff = bv_out @ Wv_in + bv_in    (fp32, [N])
// 256x256x256 fp32 — negligible cost, removes a full M x 256 x 256 GEMM.
// ---------------------------------------------------------------------------
__global__ void fold_value_weights(const float* __restrict__ Wvo,
                                   const float* __restrict__ bvo,
                                   const float* __restrict__ Wvi,
                                   const float* __restrict__ bvi,
                                   __bf16* __restrict__ WT,
                                   float*  __restrict__ beff) {
    int i = blockIdx.x * blockDim.x + threadIdx.x;
    if (i >= EMB * EMB) return;
    int k = i / EMB;
    int n = i - k * EMB;
    float s = 0.0f;
    for (int j = 0; j < EMB; ++j) s += Wvo[k * EMB + j] * Wvi[j * EMB + n];
    WT[n * EMB + k] = (__bf16)s;
    if (k == 0) {
        float b = bvi[n];
        for (int j = 0; j < EMB; ++j) b += bvo[j] * Wvi[j * EMB + n];
        beff[n] = b;
    }
}

// ---------------------------------------------------------------------------
// bf16 WMMA GEMM, K fixed at 256 (EMB), 16M x 64N per wave:
// 4 accumulators share one A fragment; two fragment sets ping-pong so each
// WMMA batch consumes loads issued two batches earlier — no register copies,
// fully unrolled K loop, partial load-waits with next loads in flight.
//   A  : bf16 row-major [M x 256]
//   BT : bf16 row-major [N x 256]  (B transposed)
// Requires N % 64 == 0, M % 16 == 0.
// ---------------------------------------------------------------------------
__global__ __launch_bounds__(256)
void gemm_bf16_wmma(const __bf16* __restrict__ A,
                    const __bf16* __restrict__ BT,
                    const float*  __restrict__ bias,
                    const float*  __restrict__ residual,
                    float*        __restrict__ outF,
                    __bf16*       __restrict__ outH,
                    int Mrows, int N) {
    constexpr int K = EMB;                          // 256, compile-time
    const int lane   = threadIdx.x;                 // 0..31
    const int tilesN = N >> 6;                      // 64-wide N tiles
    const int tiles  = (Mrows >> 4) * tilesN;
    const int t = blockIdx.x * blockDim.y + threadIdx.y;   // wave-uniform
    if (t >= tiles) return;                         // uniform exit: EXEC all-1 for WMMA

    const int tileM = t / tilesN;
    const int tileN = (t - tileM * tilesN) * 4;     // in 16-column units

    // A fragment: lane (0-15) = row M, K chunks [kA..kA+7] and [kA+16..kA+23]
    const int rowA = tileM * 16 + (lane & 15);
    const int kA   = (lane >> 4) * 8;
    // B fragment: lane (0-15) = col N, K chunks [kB..kB+7] and [kB+16..kB+23]
    const int kB   = (lane >> 4) * 8;

    const __bf16* aptr  = A  + (long)rowA * K + kA;
    const __bf16* bptr0 = BT + (long)(tileN * 16 + (lane & 15)) * K + kB;

    auto loadfrags = [&](int k, ABfrag& a, ABfrag* b) {
        a.u[0] = *(const u32x4*)(aptr + k);
        a.u[1] = *(const u32x4*)(aptr + k + 16);
#pragma unroll
        for (int j = 0; j < 4; ++j) {
            const __bf16* bp = bptr0 + (long)j * 16 * K + k;
            b[j].u[0] = *(const u32x4*)(bp);
            b[j].u[1] = *(const u32x4*)(bp + 16);
        }
    };

    v8f acc[4] = {};
    auto wmma4 = [&](const ABfrag& a, const ABfrag* b) {
#pragma unroll
        for (int j = 0; j < 4; ++j) {
            acc[j] = __builtin_amdgcn_wmma_f32_16x16x32_bf16(
                /*neg_a=*/false, a.v, /*neg_b=*/false, b[j].v,
                /*c_mod=*/(short)0, acc[j], /*reuse_a=*/false, /*reuse_b=*/false);
        }
    };

    // Ping-pong double buffering; K/32 = 8 steps (even), fully unrolled.
    ABfrag a0, b0[4], a1, b1[4];
    loadfrags(0, a0, b0);
    loadfrags(32, a1, b1);
#pragma unroll
    for (int k = 0; k < K; k += 64) {
        wmma4(a0, b0);
        if (k + 64 < K) loadfrags(k + 64, a0, b0);   // compile-time resolved
        wmma4(a1, b1);
        if (k + 96 < K) loadfrags(k + 96, a1, b1);   // compile-time resolved
    }

    // Epilogue: C/D layout -> VGPR r holds rows (r, r+8), lane&15 = column
    const int mbase = tileM * 16 + ((lane >> 4) << 3);
#pragma unroll
    for (int j = 0; j < 4; ++j) {
        const int n  = (tileN + j) * 16 + (lane & 15);
        const float bv = bias ? bias[n] : 0.0f;
#pragma unroll
        for (int r = 0; r < 8; ++r) {
            const long o = (long)(mbase + r) * N + n;
            float val = acc[j][r] + bv;
            if (residual) val += residual[o];
            if (outF) outF[o] = val;
            if (outH) outH[o] = (__bf16)val;
        }
    }
}

// ---------------------------------------------------------------------------
// Deformable sampling: one wave per (b, l, head); lane = channel d (0..31).
// v image is (H=L, W=1): grid-x reduces to an in-bounds weight; grid-y maps to
// rows centered at L/2. Softmax over the 8 points fused in.
//   v    : fp32 [B*L, 256]   (channel = h*32 + d)
//   off  : fp32 [B*L, 128]   (h*16 + p*2 + {x,y})
//   awl  : fp32 [B*L, 64]    (h*8 + p) softmax logits
//   attn : bf16 [B*L, 256]   output (input to next GEMM)
// ---------------------------------------------------------------------------
__global__ __launch_bounds__(256)
void deform_sample(const float* __restrict__ v,
                   const float* __restrict__ off,
                   const float* __restrict__ awl,
                   __bf16* __restrict__ attn,
                   int BL, int Lrows) {
    const int lane = threadIdx.x;                        // channel d
    const int g = blockIdx.x * blockDim.y + threadIdx.y; // (bl, h) pair
    if (g >= BL * NH) return;
    const int h  = g & (NH - 1);
    const int bl = g >> 3;
    const int b  = bl / Lrows;
    const int rowbase = b * Lrows;

    // softmax over the 8 per-point logits (broadcast loads, redundant per lane)
    const float* lg = awl + (long)bl * (NH * NP) + h * NP;
    float l[NP];
    float mx = -INFINITY;
#pragma unroll
    for (int p = 0; p < NP; ++p) { l[p] = lg[p]; mx = fmaxf(mx, l[p]); }
    float s = 0.0f;
#pragma unroll
    for (int p = 0; p < NP; ++p) { l[p] = __expf(l[p] - mx); s += l[p]; }
    const float inv = 1.0f / s;

    const float* op = off + (long)bl * (NH * NP * 2) + h * (NP * 2);
    float acc = 0.0f;
#pragma unroll
    for (int p = 0; p < NP; ++p) {
        const float ox = op[2 * p + 0];
        const float oy = op[2 * p + 1];
        const float aw = l[p] * inv;

        // x axis: W == 1 -> pixel gx = ox; only cell 0 is in-bounds
        const float x0f = floorf(ox);
        const float wx1 = ox - x0f;
        float wx = 0.0f;
        if (x0f == 0.0f)  wx = 1.0f - wx1;   // corner x0   valid
        if (x0f == -1.0f) wx = wx1;          // corner x0+1 valid
        // y axis: pixel gy = oy + L/2 - 0.5
        const float gy  = oy + 0.5f * (float)Lrows - 0.5f;
        const float y0f = floorf(gy);
        const float wy1 = gy - y0f;
        const int   y0  = (int)y0f;

        const float w0 = aw * wx * (1.0f - wy1);
        const float w1 = aw * wx * wy1;
        if (y0 >= 0 && y0 < Lrows)
            acc += w0 * v[(long)(rowbase + y0) * EMB + h * DH + lane];
        if (y0 + 1 >= 0 && y0 + 1 < Lrows)
            acc += w1 * v[(long)(rowbase + y0 + 1) * EMB + h * DH + lane];
    }
    attn[(long)bl * EMB + h * DH + lane] = (__bf16)acc;
}

// ---------------------------------------------------------------------------
// Host-side orchestration
// ---------------------------------------------------------------------------
static inline size_t align_up(size_t v, size_t a) { return (v + a - 1) & ~(a - 1); }

extern "C" void kernel_launch(void* const* d_in, const int* in_sizes, int n_in,
                              void* d_out, int out_size, void* d_ws, size_t ws_size,
                              hipStream_t stream) {
    (void)in_sizes; (void)n_in; (void)out_size; (void)ws_size;

    const float* x      = (const float*)d_in[0];
    const float* Wv_out = (const float*)d_in[1];
    const float* bv_out = (const float*)d_in[2];
    const float* Wv_in  = (const float*)d_in[3];
    const float* bv_in  = (const float*)d_in[4];
    const float* Wso    = (const float*)d_in[5];
    const float* bso    = (const float*)d_in[6];
    const float* Waw    = (const float*)d_in[7];
    const float* baw    = (const float*)d_in[8];
    const float* Wo_in  = (const float*)d_in[9];
    const float* bo_in  = (const float*)d_in[10];
    const float* Wo_out = (const float*)d_in[11];
    const float* bo_out = (const float*)d_in[12];
    float* out = (float*)d_out;

    // ---- workspace carve-out ----
    char* cur = (char*)d_ws;
    auto take = [&](size_t bytes) {
        char* p = cur;
        cur += align_up(bytes, 256);
        return p;
    };
    __bf16* x_h     = (__bf16*)take((size_t)M * EMB * 2);       // x in bf16
    float*  v_f     = (float*) take((size_t)M * EMB * 4);       // fused value proj (gather src)
    float*  off_f   = (float*) take((size_t)M * NH * NP * 2 * 4);
    float*  aw_f    = (float*) take((size_t)M * NH * NP * 4);
    __bf16* attn_h  = (__bf16*)take((size_t)M * EMB * 2);       // sampled output
    __bf16* y_h     = (__bf16*)take((size_t)M * EMB * 2);       // inner out-proj + residual
    __bf16* WveT    = (__bf16*)take((size_t)EMB * EMB * 2);     // folded value weight (T)
    float*  bve     = (float*) take((size_t)EMB * 4);           // folded value bias
    __bf16* WsoT    = (__bf16*)take((size_t)EMB * (NH * NP * 2) * 2);
    __bf16* WawT    = (__bf16*)take((size_t)EMB * (NH * NP) * 2);
    __bf16* WoiT    = (__bf16*)take((size_t)EMB * EMB * 2);
    __bf16* WooT    = (__bf16*)take((size_t)EMB * EMB * 2);

    // ---- prep: activations + weights ----
    {
        int n = M * EMB;
        cvt_f32_to_bf16<<<(n + 255) / 256, 256, 0, stream>>>(x, x_h, n);
    }
    fold_value_weights<<<(EMB * EMB + 255) / 256, 256, 0, stream>>>(
        Wv_out, bv_out, Wv_in, bv_in, WveT, bve);
    auto tw = [&](const float* W, __bf16* WT, int K, int N) {
        int n = K * N;
        transpose_cvt<<<(n + 255) / 256, 256, 0, stream>>>(W, WT, K, N);
    };
    tw(Wso,    WsoT, EMB, NH * NP * 2);
    tw(Waw,    WawT, EMB, NH * NP);
    tw(Wo_in,  WoiT, EMB, EMB);
    tw(Wo_out, WooT, EMB, EMB);

    // ---- GEMM launcher (16M x 64N per wave, K = 256) ----
    dim3 blk(32, 8);
    auto gemm = [&](const __bf16* A, const __bf16* BT, const float* bias,
                    const float* res, float* oF, __bf16* oH, int N) {
        int tiles = (M / 16) * (N / 64);
        int grid = (tiles + 7) / 8;
        gemm_bf16_wmma<<<grid, blk, 0, stream>>>(A, BT, bias, res, oF, oH, M, N);
    };

    // 1) v = x @ (Wv_out@Wv_in) + folded bias       (fp32 gather source)
    gemm(x_h, WveT, bve, nullptr, v_f, nullptr, EMB);
    // 2) sampling offsets = x @ Wso + bso           (fp32)
    gemm(x_h, WsoT, bso, nullptr, off_f, nullptr, NH * NP * 2);
    // 3) attention logits = x @ Waw + baw           (fp32)
    gemm(x_h, WawT, baw, nullptr, aw_f, nullptr, NH * NP);

    // 4) softmax + bilinear gather + point-weighted sum
    {
        int waves = M * NH;           // one wave per (b, l, head)
        int grid = (waves + 7) / 8;
        deform_sample<<<grid, blk, 0, stream>>>(v_f, off_f, aw_f, attn_h, M, LQ);
    }

    // 5) y = attn @ Wo_in + bo_in + x (residual)    (bf16 for final GEMM)
    gemm(attn_h, WoiT, bo_in, x, nullptr, y_h, EMB);
    // 6) out = y @ Wo_out + bo_out                  (fp32 final output)
    gemm(y_h, WooT, bo_out, nullptr, out, nullptr, EMB);
}
